// TransformerBlock_52527450030541
// MI455X (gfx1250) — matrix operand order repaired and checked by reading, not compiled, not run on hardware
//
#include <hip/hip_runtime.h>

// ---------------- problem constants ----------------
#define B_      2
#define SEQ_    2048
#define DIM_    4096
#define NH_     32
#define NKV_    8
#define HEAD_   128
#define NREP_   4
#define HIDDEN_ 11008
#define TOK_    (B_*SEQ_)
#define KVDIM_  (NKV_*HEAD_)   // 1024

typedef __attribute__((ext_vector_type(16))) __bf16        v16bf;
typedef __attribute__((ext_vector_type(8)))  float         v8f;
typedef __attribute__((ext_vector_type(4)))  unsigned int  u32x4;
typedef __attribute__((ext_vector_type(8)))  int           i32x8;
typedef __attribute__((ext_vector_type(4)))  int           i32x4;

#ifndef __has_builtin
#define __has_builtin(x) 0
#endif
#if defined(__HIP_DEVICE_COMPILE__) && __has_builtin(__builtin_amdgcn_tensor_load_to_lds)
#define HAVE_TDM 1
#else
#define HAVE_TDM 0
#endif

union Frag { v16bf v; u32x4 u[2]; };

__device__ __forceinline__ unsigned short f2bf(float f) {
  unsigned u = __float_as_uint(f);
  return (unsigned short)((u + 0x7FFFu + ((u >> 16) & 1u)) >> 16);  // RNE
}
__device__ __forceinline__ float bf2f(unsigned short h) {
  return __uint_as_float(((unsigned)h) << 16);
}

#if HAVE_TDM
// Issue a 2D TENSOR_LOAD_TO_LDS: tile (tile_d0 x tile_d1) elements of 2 bytes,
// row stride stride0 elements, LDS padding pad_amount after each pad_interval.
// D# bitfields per CDNA5 ISA ch.8 (group0 128b, group1 256b; groups 2/3 zero).
__device__ __forceinline__ void tdm_load_2d(unsigned ldsOff, const void* gaddr,
                                            unsigned tensor_d0, unsigned tensor_d1,
                                            unsigned tile_d0, unsigned tile_d1,
                                            unsigned stride0,
                                            unsigned pad_int_code, unsigned pad_amt_code) {
  unsigned long long ga = (unsigned long long)gaddr;
  u32x4 g0;
  g0[0] = 1u;                                            // count=1 (valid), user mode
  g0[1] = ldsOff;                                        // lds_addr (bytes)
  g0[2] = (unsigned)ga;                                  // global_addr[31:0]
  g0[3] = (unsigned)((ga >> 32) & 0x1FFFFFFu) | 0x80000000u; // addr[56:32] | type=2
  i32x8 g1;
  g1[0] = (int)((1u << 16)                               // data_size = 2B
              | (1u << 20)                               // pad_enable
              | (pad_int_code << 22) | (pad_amt_code << 25));
  g1[1] = (int)((tensor_d0 & 0xFFFFu) << 16);            // tensor_dim0[15:0]
  g1[2] = (int)((tensor_d0 >> 16) | ((tensor_d1 & 0xFFFFu) << 16));
  g1[3] = (int)((tensor_d1 >> 16) | (tile_d0 << 16));    // tile_dim0
  g1[4] = (int)(tile_d1 & 0xFFFFu);                      // tile_dim1 (tile_dim2=0)
  g1[5] = (int)stride0;                                  // tensor_dim0_stride[31:0]
  g1[6] = 0;
  g1[7] = 0;
  i32x4 z4 = {0, 0, 0, 0};
#if __clang_major__ >= 23
  i32x8 z8 = {0, 0, 0, 0, 0, 0, 0, 0};
  __builtin_amdgcn_tensor_load_to_lds(g0, g1, z4, z4, z8, 0);
#else
  __builtin_amdgcn_tensor_load_to_lds(g0, g1, z4, z4, 0);
#endif
}
#endif

// ---------------- elementwise converts ----------------
__global__ __launch_bounds__(256) void f32_to_bf16(const float* __restrict__ in,
                                                   unsigned short* __restrict__ outp,
                                                   size_t n) {
  size_t i = (size_t)blockIdx.x * 256 + threadIdx.x;
  if (i < n) outp[i] = f2bf(in[i]);
}

// W [K][N] f32  ->  Wt [N][K] bf16  (tiled transpose)
__global__ void transpose_convert(const float* __restrict__ W,
                                  unsigned short* __restrict__ Wt,
                                  int K, int N) {
  __shared__ float tile[32][33];
  int bx = blockIdx.x * 32;  // N offset
  int by = blockIdx.y * 32;  // K offset
  int tx = threadIdx.x, ty = threadIdx.y;
#pragma unroll
  for (int i = 0; i < 32; i += 8)
    tile[ty + i][tx] = W[(size_t)(by + ty + i) * N + bx + tx];
  __syncthreads();
#pragma unroll
  for (int i = 0; i < 32; i += 8)
    Wt[(size_t)(bx + ty + i) * K + by + tx] = f2bf(tile[tx][ty + i]);
}

// interleaved RoPE, in place on bf16 [tok][nheads][HEAD]
__global__ __launch_bounds__(256) void rope_kernel(unsigned short* __restrict__ t,
                                                   const float* __restrict__ cs,
                                                   const float* __restrict__ sn,
                                                   int nheads, size_t npairs) {
  size_t i = (size_t)blockIdx.x * 256 + threadIdx.x;
  if (i >= npairs) return;
  int pair  = (int)(i & 63);            // HEAD/2 = 64
  size_t th = i >> 6;
  int hh    = (int)(th % nheads);
  size_t tok = th / nheads;
  int s = (int)(tok % SEQ_);
  float c = cs[s * 64 + pair], sv = sn[s * 64 + pair];
  size_t base = (tok * (size_t)nheads + hh) * HEAD_ + (size_t)pair * 2;
  float t0 = bf2f(t[base]), t1 = bf2f(t[base + 1]);
  t[base]     = f2bf(t0 * c - t1 * sv);
  t[base + 1] = f2bf(t0 * sv + t1 * c);
}

// v [b][s][kv][d] bf16 -> vt [b][kv][d][s] bf16
__global__ __launch_bounds__(256) void vtrans_kernel(const unsigned short* __restrict__ v,
                                                     unsigned short* __restrict__ vt,
                                                     size_t n) {
  size_t i = (size_t)blockIdx.x * 256 + threadIdx.x;
  if (i >= n) return;
  int s = (int)(i % SEQ_);
  size_t r = i / SEQ_;
  int d  = (int)(r % HEAD_);
  size_t r2 = r / HEAD_;
  int kv = (int)(r2 % NKV_);
  int b  = (int)(r2 / NKV_);
  vt[i] = v[(((size_t)b * SEQ_ + s) * NKV_ + kv) * HEAD_ + d];
}

// ---------------- bf16 WMMA GEMM ----------------
// C[M][N] = A[M][K] * Bt[N][K]^T, BK=64 (16 WMMAs per barrier pair), epilogues:
//   mode 0: Cf = acc; mode 1: Cf = acc + resid; mode 2: Cb = bf16(silu(gate)*acc);
//   mode 3: Cb = bf16(acc)
#define LDA_ 72   // bf16 elements per LDS row (64 data + 16B pad): conflict-free b128
__global__ __launch_bounds__(256) void gemm_bf16_wmma(
    const unsigned short* __restrict__ A,
    const unsigned short* __restrict__ Bt,
    float* __restrict__ Cf, unsigned short* __restrict__ Cb,
    const float* __restrict__ resid, const float* __restrict__ gate,
    int M, int N, int K, int mode) {
  __shared__ unsigned short Al[128 * LDA_];   // 18432 B
  __shared__ unsigned short Bl[128 * LDA_];   // 18432 B
  const int tid  = threadIdx.x;
  const int wave = tid >> 5, lane = tid & 31;
  const int lhalf = lane >> 4, r16 = lane & 15;
  const int wm = wave >> 2, wn = wave & 3;          // 2x4 wave grid, wave tile 64x32
  const long bm = (long)blockIdx.y * 128, bn = (long)blockIdx.x * 128;

  const v8f vzero = {0.f,0.f,0.f,0.f,0.f,0.f,0.f,0.f};
  v8f acc[4][2];
#pragma unroll
  for (int i = 0; i < 4; i++)
#pragma unroll
    for (int j = 0; j < 2; j++) acc[i][j] = vzero;

#if HAVE_TDM
  const unsigned AlOff = (unsigned)(unsigned long long)(void*)Al;
  const unsigned BlOff = (unsigned)(unsigned long long)(void*)Bl;
#endif

  for (int k0 = 0; k0 < K; k0 += 64) {
#if HAVE_TDM
    // Tensor Data Mover: one wave DMAs both 128x64 bf16 tiles straight to LDS
    // (pad 16B per 128B row -> LDS stride LDA_), tracked by TENSORcnt.
    if (wave == 0) {
      tdm_load_2d(AlOff, (const void*)(A + bm * (long)K + k0),
                  (unsigned)(K - k0), (unsigned)(M - (int)bm),
                  64u, 128u, (unsigned)K, 4u, 3u);
      tdm_load_2d(BlOff, (const void*)(Bt + bn * (long)K + k0),
                  (unsigned)(K - k0), (unsigned)(N - (int)bn),
                  64u, 128u, (unsigned)K, 4u, 3u);
#if __has_builtin(__builtin_amdgcn_s_wait_tensorcnt)
      __builtin_amdgcn_s_wait_tensorcnt(0);
#else
      asm volatile("s_wait_tensorcnt 0x0" ::: "memory");
#endif
    }
    asm volatile("" ::: "memory");  // LDS now holds TDM-written tiles
#else
#pragma unroll
    for (int p = 0; p < 4; ++p) {                   // 128x64 bf16 tile = 1024 x b128
      int idx = tid + p * 256;
      int r = idx >> 3, c = idx & 7;
      *(u32x4*)(Al + r * LDA_ + c * 8) = *(const u32x4*)(A  + (bm + r) * (long)K + k0 + c * 8);
      *(u32x4*)(Bl + r * LDA_ + c * 8) = *(const u32x4*)(Bt + (bn + r) * (long)K + k0 + c * 8);
    }
#endif
    if (k0 + 64 < K) {                              // WGP-scope prefetch of next K-slice
      __builtin_prefetch(A  + (bm + (tid >> 3)) * (long)K + k0 + 64, 0, 3);
      __builtin_prefetch(Bt + (bn + (tid >> 3)) * (long)K + k0 + 64, 0, 3);
    }
    __syncthreads();

#pragma unroll
    for (int ks = 0; ks < 2; ++ks) {
      Frag af[4], bfr[2];
#pragma unroll
      for (int i = 0; i < 4; i++) {                 // A frag: lane=row, K chunks {h*8, 16+h*8}
        int r = wm * 64 + i * 16 + r16;
        af[i].u[0] = *(const u32x4*)(Al + r * LDA_ + ks * 32 + lhalf * 8);
        af[i].u[1] = *(const u32x4*)(Al + r * LDA_ + ks * 32 + 16 + lhalf * 8);
      }
#pragma unroll
      for (int j = 0; j < 2; j++) {                 // B frag: lane=col, K chunks {h*8, 16+h*8}
        int r = wn * 32 + j * 16 + r16;
        bfr[j].u[0] = *(const u32x4*)(Bl + r * LDA_ + ks * 32 + lhalf * 8);
        bfr[j].u[1] = *(const u32x4*)(Bl + r * LDA_ + ks * 32 + 16 + lhalf * 8);
      }
#pragma unroll
      for (int i = 0; i < 4; i++)
#pragma unroll
        for (int j = 0; j < 2; j++)
          acc[i][j] = __builtin_amdgcn_wmma_f32_16x16x32_bf16(
              false, af[i].v, false, bfr[j].v, (short)0, acc[i][j], false, false);
    }
    __syncthreads();
  }

#pragma unroll
  for (int i = 0; i < 4; i++) {
    long gm0 = bm + wm * 64 + i * 16 + lhalf * 8;   // C layout: reg e -> m = e + 8*half
#pragma unroll
    for (int j = 0; j < 2; j++) {
      long gn = bn + wn * 32 + j * 16 + r16;
#pragma unroll
      for (int e = 0; e < 8; e++) {
        long off = (gm0 + e) * (long)N + gn;
        float v = acc[i][j][e];
        if (mode == 0)      Cf[off] = v;
        else if (mode == 1) Cf[off] = v + resid[off];
        else if (mode == 2) { float g = gate[off]; Cb[off] = f2bf((g / (1.0f + __expf(-g))) * v); }
        else                Cb[off] = f2bf(v);
      }
    }
  }
}

// ---------------- flash attention (causal, GQA) ----------------
// block: 128 q-rows (8 waves x 16 rows), Bc = 32 keys per iteration
__global__ __launch_bounds__(256) void flash_attn_wmma(
    const unsigned short* __restrict__ qb,   // [tok][NH][HEAD] bf16 (roped)
    const unsigned short* __restrict__ kb,   // [tok][NKV][HEAD] bf16 (roped)
    const unsigned short* __restrict__ vtb,  // [b][kv][d][s] bf16
    unsigned short* __restrict__ ob) {       // [tok][NH][HEAD] bf16
  __shared__ unsigned short Qs[128 * 136];   // 34816 B
  __shared__ unsigned short Ks[32 * 136];    //  8704 B
  __shared__ unsigned short Vs[128 * 40];    // 10240 B
  __shared__ unsigned short Ps[128 * 40];    // 10240 B  (total 64000 B)

  const int tid  = threadIdx.x;
  const int wave = tid >> 5, lane = tid & 31;
  const int lhalf = lane >> 4, r16 = lane & 15;
  const int qt = blockIdx.x;
  const int bh = blockIdx.y;
  const int b  = bh / NH_, h = bh % NH_;
  const int kv = h / NREP_;
  const int q0 = qt * 128;

#pragma unroll
  for (int pss = 0; pss < 8; ++pss) {        // Q tile 128x128 bf16
    int idx = tid + pss * 256;
    int r = idx >> 4, c = idx & 15;
    size_t g = ((size_t)(b * SEQ_ + q0 + r) * NH_ + h) * HEAD_ + c * 8;
    *(u32x4*)(Qs + r * 136 + c * 8) = *(const u32x4*)(qb + g);
  }
  __syncthreads();

  const v8f vzero = {0.f,0.f,0.f,0.f,0.f,0.f,0.f,0.f};
  float m_run[8], l_run[8];
  v8f o[8];
#pragma unroll
  for (int e = 0; e < 8; ++e) { m_run[e] = -3.0e38f; l_run[e] = 0.0f; o[e] = vzero; }

  const float scale = 0.088388347648318447f; // 1/sqrt(128)
  const int kt_end = (q0 + 127) >> 5;        // inclusive causal bound
  for (int kt = 0; kt <= kt_end; ++kt) {
    const int k0 = kt * 32;
#pragma unroll
    for (int pss = 0; pss < 2; ++pss) {      // K tile 32x128
      int idx = tid + pss * 256;
      int r = idx >> 4, c = idx & 15;
      size_t g = ((size_t)(b * SEQ_ + k0 + r) * NKV_ + kv) * HEAD_ + c * 8;
      *(u32x4*)(Ks + r * 136 + c * 8) = *(const u32x4*)(kb + g);
    }
#pragma unroll
    for (int pss = 0; pss < 2; ++pss) {      // V^T tile 128x32
      int idx = tid + pss * 256;
      int r = idx >> 2, c = idx & 3;
      size_t g = ((size_t)((b * NKV_ + kv) * HEAD_ + r)) * SEQ_ + k0 + c * 8;
      *(u32x4*)(Vs + r * 40 + c * 8) = *(const u32x4*)(vtb + g);
    }
    __syncthreads();

    // S = Q K^T  : 16 rows x 32 cols per wave
    v8f s[2]; s[0] = vzero; s[1] = vzero;
#pragma unroll
    for (int ks = 0; ks < 4; ++ks) {         // HEAD/32
      Frag af;
      int ar = wave * 16 + r16;
      af.u[0] = *(const u32x4*)(Qs + ar * 136 + ks * 32 + lhalf * 8);
      af.u[1] = *(const u32x4*)(Qs + ar * 136 + ks * 32 + 16 + lhalf * 8);
#pragma unroll
      for (int j = 0; j < 2; ++j) {
        Frag kf;
        int br = j * 16 + r16;
        kf.u[0] = *(const u32x4*)(Ks + br * 136 + ks * 32 + lhalf * 8);
        kf.u[1] = *(const u32x4*)(Ks + br * 136 + ks * 32 + 16 + lhalf * 8);
        s[j] = __builtin_amdgcn_wmma_f32_16x16x32_bf16(
            false, af.v, false, kf.v, (short)0, s[j], false, false);
      }
    }

    // streaming softmax (row = e + 8*lhalf within the wave's 16 rows)
#pragma unroll
    for (int e = 0; e < 8; ++e) {
      int qi = q0 + wave * 16 + e + lhalf * 8;
      float mx = -3.0e38f;
#pragma unroll
      for (int j = 0; j < 2; ++j) {
        int kj = k0 + j * 16 + r16;
        float v = s[j][e] * scale + ((kj > qi) ? -1.0e9f : 0.0f);
        s[j][e] = v;
        mx = fmaxf(mx, v);
      }
#pragma unroll
      for (int off = 1; off < 16; off <<= 1) mx = fmaxf(mx, __shfl_xor(mx, off, 32));
      float mnew = fmaxf(m_run[e], mx);
      float acc = 0.0f;
#pragma unroll
      for (int j = 0; j < 2; ++j) {
        float pv = __expf(s[j][e] - mnew);
        s[j][e] = pv;
        acc += pv;
      }
#pragma unroll
      for (int off = 1; off < 16; off <<= 1) acc += __shfl_xor(acc, off, 32);
      float alpha = __expf(m_run[e] - mnew);
      l_run[e] = l_run[e] * alpha + acc;
      m_run[e] = mnew;
#pragma unroll
      for (int nt = 0; nt < 8; ++nt) o[nt][e] *= alpha;
    }

    // stage P (C-frag layout) -> LDS -> reload as A-frag; same-wave rows only
#pragma unroll
    for (int e = 0; e < 8; ++e) {
      int ml = wave * 16 + e + lhalf * 8;
#pragma unroll
      for (int j = 0; j < 2; ++j)
        Ps[ml * 40 + j * 16 + r16] = f2bf(s[j][e]);
    }
    Frag pf;
    int pr = wave * 16 + r16;
    pf.u[0] = *(const u32x4*)(Ps + pr * 40 + lhalf * 8);
    pf.u[1] = *(const u32x4*)(Ps + pr * 40 + 16 + lhalf * 8);
#pragma unroll
    for (int nt = 0; nt < 8; ++nt) {         // O += P @ V  (K=32, one step)
      Frag vf;
      int vr = nt * 16 + r16;
      vf.u[0] = *(const u32x4*)(Vs + vr * 40 + lhalf * 8);
      vf.u[1] = *(const u32x4*)(Vs + vr * 40 + 16 + lhalf * 8);
      o[nt] = __builtin_amdgcn_wmma_f32_16x16x32_bf16(
          false, pf.v, false, vf.v, (short)0, o[nt], false, false);
    }
    __syncthreads();
  }

#pragma unroll
  for (int e = 0; e < 8; ++e) {
    float inv = 1.0f / l_run[e];
    int qrow = q0 + wave * 16 + e + lhalf * 8;
    size_t base = ((size_t)(b * SEQ_ + qrow) * NH_ + h) * HEAD_;
#pragma unroll
    for (int nt = 0; nt < 8; ++nt)
      ob[base + nt * 16 + r16] = f2bf(o[nt][e] * inv);
  }
}

// ---------------- driver ----------------
extern "C" void kernel_launch(void* const* d_in, const int* in_sizes, int n_in,
                              void* d_out, int out_size, void* d_ws, size_t ws_size,
                              hipStream_t stream) {
  (void)in_sizes; (void)n_in; (void)out_size; (void)ws_size;
  const float* x  = (const float*)d_in[0];
  // d_in[1] = mask (unused; causal mask computed analytically)
  const float* cs = (const float*)d_in[2];
  const float* sn = (const float*)d_in[3];
  const float* wq = (const float*)d_in[4];
  const float* wk = (const float*)d_in[5];
  const float* wv = (const float*)d_in[6];
  const float* wo = (const float*)d_in[7];
  const float* w1 = (const float*)d_in[8];
  const float* w2 = (const float*)d_in[9];
  const float* w3 = (const float*)d_in[10];
  float* out = (float*)d_out;

  char* p = (char*)d_ws;
  auto alloc = [&](size_t bytes) -> char* {
    char* q = p; p += (bytes + 255) & ~(size_t)255; return q;
  };
  // ~852 MB total workspace
  unsigned short* wqT  = (unsigned short*)alloc((size_t)DIM_ * DIM_ * 2);
  unsigned short* wkT  = (unsigned short*)alloc((size_t)KVDIM_ * DIM_ * 2);
  unsigned short* wvT  = (unsigned short*)alloc((size_t)KVDIM_ * DIM_ * 2);
  unsigned short* woT  = (unsigned short*)alloc((size_t)DIM_ * DIM_ * 2);
  unsigned short* w1T  = (unsigned short*)alloc((size_t)HIDDEN_ * DIM_ * 2);
  unsigned short* w3T  = (unsigned short*)alloc((size_t)HIDDEN_ * DIM_ * 2);
  unsigned short* w2T  = (unsigned short*)alloc((size_t)DIM_ * HIDDEN_ * 2);
  unsigned short* xb   = (unsigned short*)alloc((size_t)TOK_ * DIM_ * 2);
  unsigned short* qraw = (unsigned short*)alloc((size_t)TOK_ * DIM_ * 2);
  unsigned short* kraw = (unsigned short*)alloc((size_t)TOK_ * KVDIM_ * 2);
  unsigned short* vraw = (unsigned short*)alloc((size_t)TOK_ * KVDIM_ * 2);
  unsigned short* vtb  = (unsigned short*)alloc((size_t)TOK_ * KVDIM_ * 2);
  unsigned short* attb = (unsigned short*)alloc((size_t)TOK_ * DIM_ * 2);
  float*          hf   = (float*)alloc((size_t)TOK_ * DIM_ * 4);
  unsigned short* hb   = (unsigned short*)alloc((size_t)TOK_ * DIM_ * 2);
  float*          t1   = (float*)alloc((size_t)TOK_ * HIDDEN_ * 4);
  unsigned short* gb   = (unsigned short*)alloc((size_t)TOK_ * HIDDEN_ * 2);

  dim3 tb(32, 8);
  // weights -> bf16 [N][K]
  transpose_convert<<<dim3(DIM_/32,    DIM_/32),    tb, 0, stream>>>(wq, wqT, DIM_, DIM_);
  transpose_convert<<<dim3(KVDIM_/32,  DIM_/32),    tb, 0, stream>>>(wk, wkT, DIM_, KVDIM_);
  transpose_convert<<<dim3(KVDIM_/32,  DIM_/32),    tb, 0, stream>>>(wv, wvT, DIM_, KVDIM_);
  transpose_convert<<<dim3(DIM_/32,    DIM_/32),    tb, 0, stream>>>(wo, woT, DIM_, DIM_);
  transpose_convert<<<dim3(HIDDEN_/32, DIM_/32),    tb, 0, stream>>>(w1, w1T, DIM_, HIDDEN_);
  transpose_convert<<<dim3(HIDDEN_/32, DIM_/32),    tb, 0, stream>>>(w3, w3T, DIM_, HIDDEN_);
  transpose_convert<<<dim3(DIM_/32,    HIDDEN_/32), tb, 0, stream>>>(w2, w2T, HIDDEN_, DIM_);

  // x -> bf16
  f32_to_bf16<<<(TOK_*(size_t)DIM_)/256, 256, 0, stream>>>(x, xb, (size_t)TOK_ * DIM_);

  // QKV projections (bf16 out)
  gemm_bf16_wmma<<<dim3(DIM_/128,   TOK_/128), 256, 0, stream>>>(xb, wqT, nullptr, qraw, nullptr, nullptr, TOK_, DIM_,   DIM_, 3);
  gemm_bf16_wmma<<<dim3(KVDIM_/128, TOK_/128), 256, 0, stream>>>(xb, wkT, nullptr, kraw, nullptr, nullptr, TOK_, KVDIM_, DIM_, 3);
  gemm_bf16_wmma<<<dim3(KVDIM_/128, TOK_/128), 256, 0, stream>>>(xb, wvT, nullptr, vraw, nullptr, nullptr, TOK_, KVDIM_, DIM_, 3);

  // RoPE in place; V^T for PV operand
  rope_kernel<<<((size_t)TOK_*NH_*64)/256,  256, 0, stream>>>(qraw, cs, sn, NH_,  (size_t)TOK_*NH_*64);
  rope_kernel<<<((size_t)TOK_*NKV_*64)/256, 256, 0, stream>>>(kraw, cs, sn, NKV_, (size_t)TOK_*NKV_*64);
  vtrans_kernel<<<((size_t)TOK_*KVDIM_)/256, 256, 0, stream>>>(vraw, vtb, (size_t)TOK_ * KVDIM_);

  // flash attention
  flash_attn_wmma<<<dim3(SEQ_/128, B_*NH_), 256, 0, stream>>>(qraw, kraw, vtb, attb);

  // h = x + attn @ wo
  gemm_bf16_wmma<<<dim3(DIM_/128, TOK_/128), 256, 0, stream>>>(attb, woT, hf, nullptr, x, nullptr, TOK_, DIM_, DIM_, 1);
  f32_to_bf16<<<(TOK_*(size_t)DIM_)/256, 256, 0, stream>>>(hf, hb, (size_t)TOK_ * DIM_);

  // FFN: t1 = h@w1 ; gb = silu(t1) * (h@w3) ; out = h + gb@w2
  gemm_bf16_wmma<<<dim3(HIDDEN_/128, TOK_/128), 256, 0, stream>>>(hb, w1T, t1, nullptr, nullptr, nullptr, TOK_, HIDDEN_, DIM_, 0);
  gemm_bf16_wmma<<<dim3(HIDDEN_/128, TOK_/128), 256, 0, stream>>>(hb, w3T, nullptr, gb, nullptr, t1, TOK_, HIDDEN_, DIM_, 2);
  gemm_bf16_wmma<<<dim3(DIM_/128,    TOK_/128), 256, 0, stream>>>(gb, w2T, out, nullptr, hf, nullptr, TOK_, DIM_, HIDDEN_, 1);
}
